// PairUpdate_80736795230620
// MI455X (gfx1250) — compile-verified, matrix operand order checked by reading, not run on hardware
//
#include <hip/hip_runtime.h>
#include <math.h>
#include <stdint.h>

// ---------------------------------------------------------------------------
// Sizes (B=1 fixed)
// ---------------------------------------------------------------------------
#define NSEQ 512
#define DP   128
#define DS   256
#define NN   (NSEQ * NSEQ)      // 262144 pair rows
#define XLD  136                // padded LDS row (bank-conflict avoidance), 16B-aligned rows

typedef __attribute__((ext_vector_type(16))) __bf16 bf16x16;
typedef __attribute__((ext_vector_type(8)))  __bf16 bf16x8;
typedef __attribute__((ext_vector_type(8)))  float  f32x8;

__device__ __forceinline__ float sigmoidf_(float x) { return 1.0f / (1.0f + expf(-x)); }
__device__ __forceinline__ float geluf_(float x) {
  // jax.nn.gelu default = tanh approximation
  float x3 = x * x * x;
  return 0.5f * x * (1.0f + tanhf(0.7978845608028654f * (x + 0.044715f * x3)));
}

// Load a 16x32 bf16 tile (row-major source, K contiguous, leading dim = ld elems)
// into the WMMA A/B operand layout:
//   lane l: row = l&15, K octets at koff = (l>>4)*8 : elems 0..7 -> K+0..7,
//   elems 8..15 -> K+16..23.  Works for both A (rows=M) and NT-B (rows=N).
__device__ __forceinline__ bf16x16 ld_tile(const __bf16* base, int ld, int lane) {
  const __bf16* p = base + (size_t)(lane & 15) * (size_t)ld + ((lane >> 4) << 3);
  bf16x8 lo = *(const bf16x8*)(p);
  bf16x8 hi = *(const bf16x8*)(p + 16);
  bf16x16 r;
#pragma unroll
  for (int e = 0; e < 8; ++e) { r[e] = lo[e]; r[e + 8] = hi[e]; }
  return r;
}

__device__ __forceinline__ f32x8 wmma_bf16(bf16x16 a, bf16x16 b, f32x8 c) {
  return __builtin_amdgcn_wmma_f32_16x16x32_bf16(false, a, false, b, (short)0, c,
                                                 false, false);
}

// CDNA5 async global->LDS copy of 16 bytes per lane (ASYNCcnt-tracked).
// lds_off: wave-relative LDS byte offset (low 32 bits of a generic shared ptr,
// per the ISA's flat->LDS aperture truncation rule).
__device__ __forceinline__ void async_ld_lds_b128(unsigned lds_off, const void* gaddr) {
  asm volatile("global_load_async_to_lds_b128 %0, %1, off"
               :: "v"(lds_off), "v"(gaddr) : "memory");
}
__device__ __forceinline__ void wait_asynccnt0() {
  asm volatile("s_wait_asynccnt 0x0" ::: "memory");
}

// ---------------------------------------------------------------------------
// K0: convert+transpose fp32 weight [in][out] -> bf16 [out][in]
// ---------------------------------------------------------------------------
__global__ void k_transpose_bf16(const float* __restrict__ w, __bf16* __restrict__ wt,
                                 int in_dim, int out_dim) {
  int idx = blockIdx.x * 256 + threadIdx.x;
  if (idx < in_dim * out_dim) {
    int o = idx / in_dim, i = idx - o * in_dim;
    wt[idx] = (__bf16)w[i * out_dim + o];
  }
}

// ---------------------------------------------------------------------------
// K1: sn = LN(scalar); a = sn@wa+ba ; c = sn@wb+bb      (tiny: 512 rows)
// ---------------------------------------------------------------------------
__global__ void k_scalar_proj(const float* __restrict__ sf,
                              const float* __restrict__ og, const float* __restrict__ ob,
                              const float* __restrict__ wa, const float* __restrict__ ba,
                              const float* __restrict__ wb, const float* __restrict__ bb,
                              float* __restrict__ a_out, float* __restrict__ c_out) {
  __shared__ float sn[DS];
  __shared__ float red[256];
  int i = blockIdx.x, t = threadIdx.x;
  float x = sf[(size_t)i * DS + t];
  red[t] = x;
  __syncthreads();
  for (int s = 128; s > 0; s >>= 1) { if (t < s) red[t] += red[t + s]; __syncthreads(); }
  float mu = red[0] * (1.0f / DS);
  __syncthreads();
  red[t] = x * x;
  __syncthreads();
  for (int s = 128; s > 0; s >>= 1) { if (t < s) red[t] += red[t + s]; __syncthreads(); }
  float var = red[0] * (1.0f / DS) - mu * mu;
  float inv = rsqrtf(var + 1e-5f);
  sn[t] = (x - mu) * inv * og[t] + ob[t];
  __syncthreads();
  float acc = 0.0f;
  if (t < DP) {
#pragma unroll 4
    for (int e = 0; e < DS; ++e) acc += sn[e] * wa[e * DP + t];
    a_out[(size_t)i * DP + t] = acc + ba[t];
  } else {
    int o = t - DP;
#pragma unroll 4
    for (int e = 0; e < DS; ++e) acc += sn[e] * wb[e * DP + o];
    c_out[(size_t)i * DP + o] = acc + bb[o];
  }
}

// ---------------------------------------------------------------------------
// K2: pair[i,j,d] = ((a_i ⊙ c_j) @ wo)[d] + bo[d] + pair_feats[i,j,d]
// ---------------------------------------------------------------------------
__global__ void k_pair_init(const float* __restrict__ a, const float* __restrict__ c,
                            const __bf16* __restrict__ wo_t, const float* __restrict__ bo,
                            const float* __restrict__ pf, float* __restrict__ pair) {
  int bid = blockIdx.x;
  int i = bid >> 5;
  int jb = (bid & 31) << 4;
  int tid = threadIdx.x;
  int w = tid >> 5, lane = tid & 31;

  __shared__ float   av[DP];
  __shared__ __bf16  As[16][XLD];

  if (tid < DP) av[tid] = a[(size_t)i * DP + tid];
  __syncthreads();
  for (int s = tid; s < 16 * DP; s += 256) {
    int r = s >> 7, e = s & 127;
    As[r][e] = (__bf16)(c[(size_t)(jb + r) * DP + e] * av[e]);
  }
  __syncthreads();

  int db = w * 16;
  f32x8 acc = {};
#pragma unroll
  for (int kc = 0; kc < 4; ++kc) {
    bf16x16 A = ld_tile(&As[0][0] + kc * 32, XLD, lane);
    bf16x16 B = ld_tile(wo_t + (size_t)db * DP + kc * 32, DP, lane);
    acc = wmma_bf16(A, B, acc);
  }
  int n = lane & 15, mb = (lane < 16) ? 0 : 8;
  int d = db + n;
#pragma unroll
  for (int v = 0; v < 8; ++v) {
    size_t idx = ((size_t)i * NSEQ + jb + mb + v) * DP + d;
    pair[idx] = acc[v] + bo[d] + pf[idx];
  }
}

// ---------------------------------------------------------------------------
// K3: xn = LN(pair); p = xn@wp (l,r,lg,rg), g = xn@wg; write gated/masked
// left,right in [d][row][k] bf16 (transposed per direction), gate in [row][d].
// ---------------------------------------------------------------------------
__global__ void k_tri_proj(const float* __restrict__ pair, const unsigned char* __restrict__ mask,
                           const float* __restrict__ ng, const float* __restrict__ nb,
                           const __bf16* __restrict__ wp_t, const float* __restrict__ bp,
                           const __bf16* __restrict__ wg_t, const float* __restrict__ bg,
                           __bf16* __restrict__ left_t, __bf16* __restrict__ right_t,
                           __bf16* __restrict__ gate_ws, int incoming) {
  int R0 = blockIdx.x * 16;
  int I = R0 >> 9, jb = R0 & 511;
  int tid = threadIdx.x, w = tid >> 5, lane = tid & 31;

  __shared__ __bf16 Xs[16][XLD];
  {
    int r = tid >> 4, g = tid & 15;
    const float* rowp = pair + (size_t)(R0 + r) * DP;
    float vals[8], s = 0.0f, s2 = 0.0f;
#pragma unroll
    for (int q = 0; q < 8; ++q) { float x = rowp[g + q * 16]; vals[q] = x; s += x; s2 += x * x; }
#pragma unroll
    for (int m = 8; m >= 1; m >>= 1) { s += __shfl_xor(s, m, 16); s2 += __shfl_xor(s2, m, 16); }
    float mu = s * (1.0f / DP);
    float inv = rsqrtf(s2 * (1.0f / DP) - mu * mu + 1e-5f);
#pragma unroll
    for (int q = 0; q < 8; ++q) {
      int d = g + q * 16;
      Xs[r][d] = (__bf16)((vals[q] - mu) * inv * ng[d] + nb[d]);
    }
  }
  __syncthreads();

  f32x8 accL = {}, accR = {}, accLG = {}, accRG = {}, accG = {};
  size_t cb = (size_t)(w * 16) * DP;
#pragma unroll
  for (int kc = 0; kc < 4; ++kc) {
    int kb = kc * 32;
    bf16x16 A = ld_tile(&Xs[0][0] + kb, XLD, lane);
    accL  = wmma_bf16(A, ld_tile(wp_t + cb + kb,                      DP, lane), accL);
    accR  = wmma_bf16(A, ld_tile(wp_t + cb + (size_t)128 * DP + kb,   DP, lane), accR);
    accLG = wmma_bf16(A, ld_tile(wp_t + cb + (size_t)256 * DP + kb,   DP, lane), accLG);
    accRG = wmma_bf16(A, ld_tile(wp_t + cb + (size_t)384 * DP + kb,   DP, lane), accRG);
    accG  = wmma_bf16(A, ld_tile(wg_t + cb + kb,                      DP, lane), accG);
  }

  int n = lane & 15, mb = (lane < 16) ? 0 : 8;
  int dg = w * 16 + n;
  bf16x8 lpk, rpk;
#pragma unroll
  for (int v = 0; v < 8; ++v) {
    int j = jb + mb + v;
    float mk = mask[(size_t)I * NSEQ + j] ? 1.0f : 0.0f;
    float lv = (accL[v] + bp[dg])        * sigmoidf_(accLG[v] + bp[256 + dg]) * mk;
    float rv = (accR[v] + bp[128 + dg])  * sigmoidf_(accRG[v] + bp[384 + dg]) * mk;
    gate_ws[((size_t)I * NSEQ + j) * DP + dg] = (__bf16)sigmoidf_(accG[v] + bg[dg]);
    if (!incoming) {
      lpk[v] = (__bf16)lv; rpk[v] = (__bf16)rv;
    } else {
      // incoming contracts over first pair index: store transposed [d][j][I]
      size_t idx = (size_t)dg * NN + (size_t)j * NSEQ + I;
      left_t[idx]  = (__bf16)lv;
      right_t[idx] = (__bf16)rv;
    }
  }
  if (!incoming) {
    // outgoing: [d][I][j]; 8 consecutive j per lane -> one 16B store each
    size_t base = (size_t)dg * NN + (size_t)I * NSEQ + jb + mb;
    *(bf16x8*)(left_t + base)  = lpk;
    *(bf16x8*)(right_t + base) = rpk;
  }
}

// ---------------------------------------------------------------------------
// K4: batched-over-d NT GEMM: tri_raw[d][i][j] = sum_k L[d][i][k] * R[d][j][k]
// block = (d, 128-row M strip, 64-col N strip); 8 waves, each 16x64 via 4 accs.
// B chunk (64x32) staged in LDS with double-buffered async global->LDS copies
// (ASYNCcnt), overlapped with the 4-WMMA compute of the previous chunk.
// d slowest in grid order -> per-d working set stays L2-resident.
// ---------------------------------------------------------------------------
__global__ void k_tri_einsum(const __bf16* __restrict__ left_t,
                             const __bf16* __restrict__ right_t,
                             float* __restrict__ tri_raw) {
  int bid = blockIdx.x;               // d*32 + it*8 + jt
  int d  = bid >> 5;
  int it = (bid >> 3) & 3;            // 128-row strip
  int jt = bid & 7;                   // 64-col strip
  int tid = threadIdx.x, w = tid >> 5, lane = tid & 31;

  size_t dbase = (size_t)d * NN;
  const __bf16* Lp = left_t  + dbase + (size_t)(it * 128 + w * 16) * NSEQ;
  const __bf16* Rb = right_t + dbase + (size_t)(jt * 64) * NSEQ;

  __shared__ __bf16 Bs[2][64][40];    // 40-elem rows: 80B, 16B-aligned, few conflicts

  // async stage of one 64x32 B chunk: 256 threads x 16B
  int br = tid >> 2, bseg = tid & 3;
  const __bf16* bg_src = Rb + (size_t)br * NSEQ + bseg * 8;
  unsigned bl_off0 = (unsigned)(uintptr_t)&Bs[0][br][bseg * 8];
  unsigned bl_off1 = (unsigned)(uintptr_t)&Bs[1][br][bseg * 8];

  async_ld_lds_b128(bl_off0, bg_src);            // chunk 0 -> buf 0
  wait_asynccnt0();
  __syncthreads();

  f32x8 acc0 = {}, acc1 = {}, acc2 = {}, acc3 = {};
#pragma unroll
  for (int kc = 0; kc < 16; ++kc) {
    int buf = kc & 1;
    if (kc + 1 < 16) {                           // prefetch chunk kc+1 -> other buf
      async_ld_lds_b128((kc & 1) ? bl_off0 : bl_off1, bg_src + (kc + 1) * 32);
      __builtin_prefetch(Lp + (kc + 1) * 32 + (size_t)(lane & 15) * NSEQ, 0, 1);
    }
    bf16x16 A = ld_tile(Lp + kc * 32, NSEQ, lane);
    acc0 = wmma_bf16(A, ld_tile(&Bs[buf][ 0][0], 40, lane), acc0);
    acc1 = wmma_bf16(A, ld_tile(&Bs[buf][16][0], 40, lane), acc1);
    acc2 = wmma_bf16(A, ld_tile(&Bs[buf][32][0], 40, lane), acc2);
    acc3 = wmma_bf16(A, ld_tile(&Bs[buf][48][0], 40, lane), acc3);
    wait_asynccnt0();
    __syncthreads();
  }

  int n = lane & 15, mb = (lane < 16) ? 0 : 8;
  float* outp = tri_raw + dbase + (size_t)(it * 128 + w * 16) * NSEQ + jt * 64 + n;
#pragma unroll
  for (int v = 0; v < 8; ++v) {
    size_t ro = (size_t)(mb + v) * NSEQ;
    outp[ro]      = acc0[v];
    outp[ro + 16] = acc1[v];
    outp[ro + 32] = acc2[v];
    outp[ro + 48] = acc3[v];
  }
}

// ---------------------------------------------------------------------------
// K5: pair += (LN(tri_raw) @ wo + bo) * gate     (in-place residual)
// ---------------------------------------------------------------------------
__global__ void k_tri_epilogue(const float* __restrict__ tri_raw,
                               const __bf16* __restrict__ gate_ws,
                               const float* __restrict__ hg, const float* __restrict__ hb,
                               const __bf16* __restrict__ wo_t, const float* __restrict__ bo,
                               float* __restrict__ pair) {
  int R0 = blockIdx.x * 16;
  int I = R0 >> 9, jb = R0 & 511;
  int tid = threadIdx.x, w = tid >> 5, lane = tid & 31;

  __shared__ float  Ts[16][132];
  __shared__ __bf16 Xs[16][XLD];

  for (int s = tid; s < 16 * DP; s += 256) {
    int d = s >> 4, jj = s & 15;
    Ts[jj][d] = tri_raw[(size_t)d * NN + (size_t)I * NSEQ + jb + jj];
  }
  __syncthreads();
  {
    int r = tid >> 4, g = tid & 15;
    float vals[8], s = 0.0f, s2 = 0.0f;
#pragma unroll
    for (int q = 0; q < 8; ++q) { float x = Ts[r][g + q * 16]; vals[q] = x; s += x; s2 += x * x; }
#pragma unroll
    for (int m = 8; m >= 1; m >>= 1) { s += __shfl_xor(s, m, 16); s2 += __shfl_xor(s2, m, 16); }
    float mu = s * (1.0f / DP);
    float inv = rsqrtf(s2 * (1.0f / DP) - mu * mu + 1e-5f);
#pragma unroll
    for (int q = 0; q < 8; ++q) {
      int d = g + q * 16;
      Xs[r][d] = (__bf16)((vals[q] - mu) * inv * hg[d] + hb[d]);
    }
  }
  __syncthreads();

  f32x8 acc = {};
#pragma unroll
  for (int kc = 0; kc < 4; ++kc) {
    bf16x16 A = ld_tile(&Xs[0][0] + kc * 32, XLD, lane);
    bf16x16 B = ld_tile(wo_t + (size_t)(w * 16) * DP + kc * 32, DP, lane);
    acc = wmma_bf16(A, B, acc);
  }
  int n = lane & 15, mb = (lane < 16) ? 0 : 8;
  int dg = w * 16 + n;
#pragma unroll
  for (int v = 0; v < 8; ++v) {
    size_t idx = ((size_t)I * NSEQ + jb + mb + v) * DP + dg;
    float g = (float)gate_ws[idx];
    pair[idx] = (acc[v] + bo[dg]) * g + pair[idx];
  }
}

// ---------------------------------------------------------------------------
// K6: out = gelu(LN(pair) @ w1 + b1) @ w2 + b2 + pair
// ---------------------------------------------------------------------------
__global__ void k_transition(const float* __restrict__ pair,
                             const float* __restrict__ ng, const float* __restrict__ nb,
                             const __bf16* __restrict__ w1_t, const float* __restrict__ b1,
                             const __bf16* __restrict__ w2_t, const float* __restrict__ b2,
                             float* __restrict__ out) {
  int R0 = blockIdx.x * 16;
  int tid = threadIdx.x, w = tid >> 5, lane = tid & 31;

  __shared__ __bf16 Xs[16][XLD];
  __shared__ __bf16 Hs[16][264];
  {
    int r = tid >> 4, g = tid & 15;
    const float* rowp = pair + (size_t)(R0 + r) * DP;
    float vals[8], s = 0.0f, s2 = 0.0f;
#pragma unroll
    for (int q = 0; q < 8; ++q) { float x = rowp[g + q * 16]; vals[q] = x; s += x; s2 += x * x; }
#pragma unroll
    for (int m = 8; m >= 1; m >>= 1) { s += __shfl_xor(s, m, 16); s2 += __shfl_xor(s2, m, 16); }
    float mu = s * (1.0f / DP);
    float inv = rsqrtf(s2 * (1.0f / DP) - mu * mu + 1e-5f);
#pragma unroll
    for (int q = 0; q < 8; ++q) {
      int d = g + q * 16;
      Xs[r][d] = (__bf16)((vals[q] - mu) * inv * ng[d] + nb[d]);
    }
  }
  __syncthreads();

  int n = lane & 15, mb = (lane < 16) ? 0 : 8;
#pragma unroll
  for (int half = 0; half < 2; ++half) {
    int cbcol = (w + half * 8) * 16;
    f32x8 acc = {};
#pragma unroll
    for (int kc = 0; kc < 4; ++kc) {
      bf16x16 A = ld_tile(&Xs[0][0] + kc * 32, XLD, lane);
      bf16x16 B = ld_tile(w1_t + (size_t)cbcol * DP + kc * 32, DP, lane);
      acc = wmma_bf16(A, B, acc);
    }
#pragma unroll
    for (int v = 0; v < 8; ++v) {
      int col = cbcol + n;
      Hs[mb + v][col] = (__bf16)geluf_(acc[v] + b1[col]);
    }
  }
  __syncthreads();

  f32x8 acc = {};
#pragma unroll
  for (int kc = 0; kc < 8; ++kc) {
    bf16x16 A = ld_tile(&Hs[0][0] + kc * 32, 264, lane);
    bf16x16 B = ld_tile(w2_t + (size_t)(w * 16) * 256 + kc * 32, 256, lane);
    acc = wmma_bf16(A, B, acc);
  }
  int dg = w * 16 + n;
#pragma unroll
  for (int v = 0; v < 8; ++v) {
    size_t idx = (size_t)(R0 + mb + v) * DP + dg;
    out[idx] = acc[v] + b2[dg] + pair[idx];
  }
}

// ---------------------------------------------------------------------------
// Host launcher
// ---------------------------------------------------------------------------
extern "C" void kernel_launch(void* const* d_in, const int* in_sizes, int n_in,
                              void* d_out, int out_size, void* d_ws, size_t ws_size,
                              hipStream_t stream) {
  (void)in_sizes; (void)n_in; (void)out_size; (void)ws_size;

  const float* scalar_feats = (const float*)d_in[0];
  const float* pair_feats   = (const float*)d_in[1];
  const unsigned char* mask = (const unsigned char*)d_in[2];
  const float* og = (const float*)d_in[3];
  const float* ob = (const float*)d_in[4];
  const float* wa = (const float*)d_in[5];
  const float* ba = (const float*)d_in[6];
  const float* wb = (const float*)d_in[7];
  const float* bb = (const float*)d_in[8];
  const float* wo = (const float*)d_in[9];
  const float* bo = (const float*)d_in[10];
  // to_* at 11..20, ti_* at 21..30: ng,nb,wp,bp,wg,bg,hg,hb,wo,bo
  const float* t_ng = (const float*)d_in[31];
  const float* t_nb = (const float*)d_in[32];
  const float* t_w1 = (const float*)d_in[33];
  const float* t_b1 = (const float*)d_in[34];
  const float* t_w2 = (const float*)d_in[35];
  const float* t_b2 = (const float*)d_in[36];

  // ---- workspace carve-up ----
  char* p = (char*)d_ws;
  auto alloc = [&](size_t bytes) -> void* {
    void* r = (void*)p;
    p += (bytes + 255) & ~(size_t)255;
    return r;
  };
  float*  a_ws    = (float*) alloc((size_t)NSEQ * DP * 4);
  float*  c_ws    = (float*) alloc((size_t)NSEQ * DP * 4);
  float*  pair    = (float*) alloc((size_t)NN * DP * 4);      // residual stream
  float*  tri_raw = (float*) alloc((size_t)NN * DP * 4);      // [d][i][j]
  __bf16* left_t  = (__bf16*)alloc((size_t)DP * NN * 2);      // [d][row][k]
  __bf16* right_t = (__bf16*)alloc((size_t)DP * NN * 2);
  __bf16* gate_ws = (__bf16*)alloc((size_t)NN * DP * 2);      // [row][d]
  __bf16* wo_t    = (__bf16*)alloc((size_t)DP * DP * 2);
  __bf16* wp_t[2], *wg_t[2], *wo2_t[2];
  for (int t = 0; t < 2; ++t) {
    wp_t[t]  = (__bf16*)alloc((size_t)512 * DP * 2);
    wg_t[t]  = (__bf16*)alloc((size_t)DP * DP * 2);
    wo2_t[t] = (__bf16*)alloc((size_t)DP * DP * 2);
  }
  __bf16* w1_t = (__bf16*)alloc((size_t)256 * DP * 2);
  __bf16* w2_t = (__bf16*)alloc((size_t)DP * 256 * 2);

  auto xpose = [&](const float* w, __bf16* wt, int in_d, int out_d) {
    int elems = in_d * out_d;
    k_transpose_bf16<<<(elems + 255) / 256, 256, 0, stream>>>(w, wt, in_d, out_d);
  };

  // ---- weight prep (bf16, [out][in]) ----
  xpose(wo, wo_t, DP, DP);
  for (int t = 0; t < 2; ++t) {
    int o = 11 + t * 10;
    xpose((const float*)d_in[o + 2], wp_t[t],  DP, 512);  // wp
    xpose((const float*)d_in[o + 4], wg_t[t],  DP, DP);   // wg
    xpose((const float*)d_in[o + 8], wo2_t[t], DP, DP);   // wo (tri out)
  }
  xpose(t_w1, w1_t, DP, 256);
  xpose(t_w2, w2_t, 256, DP);

  // ---- pipeline ----
  k_scalar_proj<<<NSEQ, 256, 0, stream>>>(scalar_feats, og, ob, wa, ba, wb, bb,
                                          a_ws, c_ws);
  k_pair_init<<<NSEQ * 32, 256, 0, stream>>>(a_ws, c_ws, wo_t, bo, pair_feats, pair);

  for (int t = 0; t < 2; ++t) {                 // t=0 outgoing, t=1 incoming
    int o = 11 + t * 10;
    const float* ng_ = (const float*)d_in[o + 0];
    const float* nb_ = (const float*)d_in[o + 1];
    const float* bp_ = (const float*)d_in[o + 3];
    const float* bg_ = (const float*)d_in[o + 5];
    const float* hg_ = (const float*)d_in[o + 6];
    const float* hb_ = (const float*)d_in[o + 7];
    const float* bo_ = (const float*)d_in[o + 9];

    k_tri_proj<<<NN / 16, 256, 0, stream>>>(pair, mask, ng_, nb_, wp_t[t], bp_,
                                            wg_t[t], bg_, left_t, right_t, gate_ws, t);
    k_tri_einsum<<<DP * 32, 256, 0, stream>>>(left_t, right_t, tri_raw);
    k_tri_epilogue<<<NN / 16, 256, 0, stream>>>(tri_raw, gate_ws, hg_, hb_,
                                                wo2_t[t], bo_, pair);
  }

  k_transition<<<NN / 16, 256, 0, stream>>>(pair, t_ng, t_nb, w1_t, t_b1,
                                            w2_t, t_b2, (float*)d_out);
}
